// TAD_19542101196809
// MI455X (gfx1250) — matrix operand order repaired and checked by reading, not compiled, not run on hardware
//
#include <hip/hip_runtime.h>
#include <hip/hip_bf16.h>

#define DI __device__ __forceinline__

typedef __bf16 bf16_t;
typedef __attribute__((ext_vector_type(16))) __bf16 v16bf;
typedef __attribute__((ext_vector_type(8)))  float  v8f;

// ---------------- WMMA helpers (CDNA5 bf16 16x16x32, fp32 accum) ----------------

DI v8f wmma_bf16(v16bf a, v16bf b, v8f c) {
  return __builtin_amdgcn_wmma_f32_16x16x32_bf16(false, a, false, b, (short)0, c, false, false);
}

// A fragment: row-major [M][K] bf16. lane L: M = m0 + (L&15), hi = L>>4.
// element e: K = k0 + e + 16*hi  (contiguous 32B)
DI v16bf frag_a(const bf16_t* A, int lda, int m0, int k0, int lane) {
  int m = m0 + (lane & 15), hi = lane >> 4;
  const bf16_t* p = A + (size_t)m * lda + k0 + 16 * hi;
  v16bf f;
#pragma unroll
  for (int e = 0; e < 8; e++) { f[e] = p[e]; f[8 + e] = p[8 + e]; }
  return f;
}

// B fragment from TRANSPOSED storage Bt[N][K] bf16. lane L: N = n0 + (L&15),
// element e: K = k0 + e + 16*(L>>4)  (contiguous 32B)
DI v16bf frag_b(const bf16_t* Bt, int ldb, int n0, int k0, int lane) {
  int n = n0 + (lane & 15), hi = lane >> 4;
  const bf16_t* p = Bt + (size_t)n * ldb + k0 + 16 * hi;
  v16bf f;
#pragma unroll
  for (int e = 0; e < 16; e++) f[e] = p[e];
  return f;
}

// Stage a 32-column slice of a row-major (K x N) fp32 weight into transposed bf16 LDS [32][K]
DI void stage_wt(const float* __restrict__ W, int N, int n0, int K, bf16_t* WT, int tid) {
  for (int idx = tid; idx < 32 * K; idx += 256) {
    int k = idx >> 5, nn = idx & 31;
    WT[nn * K + k] = (bf16_t)W[(size_t)k * N + n0 + nn];
  }
}

// ---- CDNA5 async global->LDS copy (ASYNCcnt-tracked, 16B per lane op) ----
// Generic pointers to LDS carry the LDS byte offset in addr[31:0] (ISA 10.2).
DI void async_copy_b128(void* lds_dst, const void* gsrc) {
  unsigned l = (unsigned)(size_t)lds_dst;
  unsigned long long g = (unsigned long long)(size_t)gsrc;
  asm volatile("global_load_async_to_lds_b128 %0, %1, off" :: "v"(l), "v"(g) : "memory");
}
DI void async_wait0() {
  asm volatile("s_wait_asynccnt 0x0" ::: "memory");
}

DI float sigmoidf_(float x) { return 1.0f / (1.0f + __expf(-x)); }
// time-encode frequency: w[d] = 10^(-9*d/255) = exp(-d * 9*ln(10)/255)
DI float twf(int d) { return __expf((float)d * -0.0812677091644953f); }

// =====================================================================
// Kernel 1: fused message-MLP + GRU over all nodes (64-row tiles)
// =====================================================================
__launch_bounds__(256)
__global__ void k_node_update(const float* __restrict__ msgs, const float* __restrict__ mem,
                              const float* __restrict__ W1, const float* __restrict__ b1,
                              const float* __restrict__ W2, const float* __restrict__ b2,
                              const float* __restrict__ Wih, const float* __restrict__ Whh,
                              const float* __restrict__ bih, const float* __restrict__ bhh,
                              float* __restrict__ upd, int Nn) {
  extern __shared__ char smem[];
  bf16_t* Xb = (bf16_t*)smem;        // 64x512
  bf16_t* H1 = Xb + 64 * 512;        // 64x256
  bf16_t* MS = H1 + 64 * 256;        // 64x128
  bf16_t* MB = MS + 64 * 128;        // 64x256
  bf16_t* WT = MB + 64 * 256;        // 32x512 (max)
  const int tid = threadIdx.x, lane = tid & 31, wv = tid >> 5;
  const int r0 = blockIdx.x * 64;
  const int rt = (wv >> 1) * 16, ctl = (wv & 1) * 16;

  for (int idx = tid; idx < 64 * 512; idx += 256) {
    int r = idx >> 9, c = idx & 511;
    int gr = r0 + r; if (gr >= Nn) gr = Nn - 1;
    Xb[idx] = (bf16_t)msgs[(size_t)gr * 512 + c];
  }
  for (int idx = tid; idx < 64 * 256; idx += 256) {
    int r = idx >> 8, c = idx & 255;
    int gr = r0 + r; if (gr >= Nn) gr = Nn - 1;
    MB[idx] = (bf16_t)mem[(size_t)gr * 256 + c];
  }
  __syncthreads();

  // h1 = relu(X @ W1 + b1)   (64x256, K=512)
  for (int nc = 0; nc < 256; nc += 32) {
    stage_wt(W1, 256, nc, 512, WT, tid);
    __syncthreads();
    v8f acc = {};
    for (int k = 0; k < 512; k += 32)
      acc = wmma_bf16(frag_a(Xb, 512, rt, k, lane), frag_b(WT, 512, ctl, k, lane), acc);
    int col = nc + ctl + (lane & 15);
    float bb = b1[col];
    int mb = rt + 8 * (lane >> 4);
#pragma unroll
    for (int i = 0; i < 8; i++) {
      float v = acc[i] + bb;
      H1[(mb + i) * 256 + col] = (bf16_t)(v > 0.f ? v : 0.f);
    }
    __syncthreads();
  }

  // msg = h1 @ W2 + b2   (64x128, K=256)
  for (int nc = 0; nc < 128; nc += 32) {
    stage_wt(W2, 128, nc, 256, WT, tid);
    __syncthreads();
    v8f acc = {};
    for (int k = 0; k < 256; k += 32)
      acc = wmma_bf16(frag_a(H1, 256, rt, k, lane), frag_b(WT, 256, ctl, k, lane), acc);
    int col = nc + ctl + (lane & 15);
    float bb = b2[col];
    int mb = rt + 8 * (lane >> 4);
#pragma unroll
    for (int i = 0; i < 8; i++) MS[(mb + i) * 128 + col] = (bf16_t)(acc[i] + bb);
    __syncthreads();
  }

  // GRU gates, 32-column chunks
  for (int nc = 0; nc < 256; nc += 32) {
    v8f ar = {}, az = {}, ain = {}, ahn = {};
    stage_wt(Wih, 768, nc, 128, WT, tid); __syncthreads();
    for (int k = 0; k < 128; k += 32)
      ar = wmma_bf16(frag_a(MS, 128, rt, k, lane), frag_b(WT, 128, ctl, k, lane), ar);
    __syncthreads();
    stage_wt(Whh, 768, nc, 256, WT, tid); __syncthreads();
    for (int k = 0; k < 256; k += 32)
      ar = wmma_bf16(frag_a(MB, 256, rt, k, lane), frag_b(WT, 256, ctl, k, lane), ar);
    __syncthreads();
    stage_wt(Wih, 768, 256 + nc, 128, WT, tid); __syncthreads();
    for (int k = 0; k < 128; k += 32)
      az = wmma_bf16(frag_a(MS, 128, rt, k, lane), frag_b(WT, 128, ctl, k, lane), az);
    __syncthreads();
    stage_wt(Whh, 768, 256 + nc, 256, WT, tid); __syncthreads();
    for (int k = 0; k < 256; k += 32)
      az = wmma_bf16(frag_a(MB, 256, rt, k, lane), frag_b(WT, 256, ctl, k, lane), az);
    __syncthreads();
    stage_wt(Wih, 768, 512 + nc, 128, WT, tid); __syncthreads();
    for (int k = 0; k < 128; k += 32)
      ain = wmma_bf16(frag_a(MS, 128, rt, k, lane), frag_b(WT, 128, ctl, k, lane), ain);
    __syncthreads();
    stage_wt(Whh, 768, 512 + nc, 256, WT, tid); __syncthreads();
    for (int k = 0; k < 256; k += 32)
      ahn = wmma_bf16(frag_a(MB, 256, rt, k, lane), frag_b(WT, 256, ctl, k, lane), ahn);
    __syncthreads();

    int col = nc + ctl + (lane & 15);
    float b_ir = bih[col],        b_hr = bhh[col];
    float b_iz = bih[256 + col],  b_hz = bhh[256 + col];
    float b_in = bih[512 + col],  b_hn = bhh[512 + col];
    int mb = rt + 8 * (lane >> 4);
#pragma unroll
    for (int i = 0; i < 8; i++) {
      int row = r0 + mb + i;
      if (row < Nn) {
        float h = mem[(size_t)row * 256 + col];
        float r = sigmoidf_(ar[i] + b_ir + b_hr);
        float z = sigmoidf_(az[i] + b_iz + b_hz);
        float n = tanhf(ain[i] + b_in + r * (ahn[i] + b_hn));
        upd[(size_t)row * 256 + col] = (1.f - z) * n + z * h;
      }
    }
  }
}

// =====================================================================
// Kernel 2: scatter mark[nodes] = 1
// =====================================================================
__global__ void k_mark(const int* __restrict__ s, const int* __restrict__ d,
                       int* __restrict__ mark, int B) {
  int i = blockIdx.x * blockDim.x + threadIdx.x;
  if (i < B) mark[s[i]] = 1;
  else if (i < 2 * B) mark[d[i - B]] = 1;
}

// =====================================================================
// Kernel 3: attention + merge -> emb (2B rows, K=10 neighbors, H=2 heads)
// =====================================================================
__launch_bounds__(256)
__global__ void k_attention(const int* __restrict__ srcn, const int* __restrict__ dstn,
                            const int* __restrict__ src_nb, const int* __restrict__ dst_nb,
                            const float* __restrict__ etime,
                            const float* __restrict__ src_nbt, const float* __restrict__ dst_nbt,
                            const float* __restrict__ upd,
                            const float* __restrict__ Wq, const float* __restrict__ Wk,
                            const float* __restrict__ Wv,
                            const float* __restrict__ mW1, const float* __restrict__ mb1,
                            const float* __restrict__ mW2, const float* __restrict__ mb2,
                            float* __restrict__ emb, int B) {
  extern __shared__ char smem[];
  bf16_t* KA  = (bf16_t*)smem;           // 64x512 A-operand staging
  bf16_t* HM  = KA + 64 * 512;           // 64x256 hmerge
  bf16_t* WT  = HM + 64 * 256;           // 32x512
  float*  QF  = (float*)(WT + 32 * 512); // 64x256 q, later ctx
  float*  NF  = QF + 64 * 256;           // 64x256 node_feat (f32)
  float*  SC  = NF + 64 * 256;           // 64x2x10 scores
  float*  DTT = SC + 64 * 20;            // 64x10 dt
  float*  TSR = DTT + 64 * 10;           // 64 timestamps
  int*    NB  = (int*)(TSR + 64);        // 64x10 neighbor ids
  int*    ND  = NB + 64 * 10;            // 64 node ids
  const int tid = threadIdx.x, lane = tid & 31, wv = tid >> 5;
  const int g0 = blockIdx.x * 64;
  const int rt = (wv >> 1) * 16, ctl = (wv & 1) * 16;
  const int twoB = 2 * B;

  for (int r = tid; r < 64; r += 256) {
    int g = g0 + r; int node = 0; float t = 0.f;
    if (g < twoB) {
      if (g < B) { node = srcn[g]; t = etime[g]; }
      else       { node = dstn[g - B]; t = etime[g - B]; }
    }
    ND[r] = node; TSR[r] = t;
  }
  __syncthreads();
  for (int idx = tid; idx < 64 * 10; idx += 256) {
    int r = idx / 10, j = idx % 10;
    int g = g0 + r; int nb = 0; float nt = 0.f;
    if (g < twoB) {
      if (g < B) { nb = src_nb[g * 10 + j]; nt = src_nbt[g * 10 + j]; }
      else       { nb = dst_nb[(g - B) * 10 + j]; nt = dst_nbt[(g - B) * 10 + j]; }
    }
    NB[idx] = nb; DTT[idx] = TSR[r] - nt;
  }
  __syncthreads();
  for (int idx = tid; idx < 64 * 256; idx += 256)
    NF[idx] = upd[(size_t)ND[idx >> 8] * 256 + (idx & 255)];
  __syncthreads();
  // q_in = [node_feat | time_encode(0)==1]
  for (int idx = tid; idx < 64 * 512; idx += 256) {
    int r = idx >> 9, c = idx & 511;
    KA[idx] = (c < 256) ? (bf16_t)NF[r * 256 + c] : (bf16_t)1.0f;
  }
  __syncthreads();
  // q = q_in @ Wq
  for (int nc = 0; nc < 256; nc += 32) {
    stage_wt(Wq, 256, nc, 512, WT, tid); __syncthreads();
    v8f acc = {};
    for (int k = 0; k < 512; k += 32)
      acc = wmma_bf16(frag_a(KA, 512, rt, k, lane), frag_b(WT, 512, ctl, k, lane), acc);
    int col = nc + ctl + (lane & 15); int mb = rt + 8 * (lane >> 4);
#pragma unroll
    for (int i = 0; i < 8; i++) QF[(mb + i) * 256 + col] = acc[i];
    __syncthreads();
  }
  for (int idx = tid; idx < 64 * 20; idx += 256) SC[idx] = 0.f;
  __syncthreads();
  // pass 1: scores[row][head][j] = q . k_j
  for (int j = 0; j < 10; j++) {
    for (int idx = tid; idx < 64 * 512; idx += 256) {
      int r = idx >> 9, c = idx & 511;
      KA[idx] = (c < 256) ? (bf16_t)upd[(size_t)NB[r * 10 + j] * 256 + c]
                          : (bf16_t)__cosf(DTT[r * 10 + j] * twf(c - 256));
    }
    __syncthreads();
    for (int nc = 0; nc < 256; nc += 32) {
      stage_wt(Wk, 256, nc, 512, WT, tid); __syncthreads();
      v8f acc = {};
      for (int k = 0; k < 512; k += 32)
        acc = wmma_bf16(frag_a(KA, 512, rt, k, lane), frag_b(WT, 512, ctl, k, lane), acc);
      int col = nc + ctl + (lane & 15); int mb = rt + 8 * (lane >> 4);
      int head = col >> 7;
#pragma unroll
      for (int i = 0; i < 8; i++) {
        float p = acc[i] * QF[(mb + i) * 256 + col];
        p += __shfl_xor(p, 1, 16); p += __shfl_xor(p, 2, 16);
        p += __shfl_xor(p, 4, 16); p += __shfl_xor(p, 8, 16);
        if ((lane & 15) == 0) atomicAdd(&SC[(mb + i) * 20 + head * 10 + j], p);
      }
      __syncthreads();
    }
  }
  // softmax over the 10 neighbors per (row, head), with pad mask (nbr==0)
  if (tid < 128) {
    int r = tid >> 1, h = tid & 1;
    float s[10]; float mx = -1e30f;
    for (int j = 0; j < 10; j++) {
      float v = SC[r * 20 + h * 10 + j] * 0.08838834764831845f; // 1/sqrt(128)
      if (NB[r * 10 + j] == 0) v = -1e9f;
      s[j] = v; mx = fmaxf(mx, v);
    }
    float sum = 0.f;
    for (int j = 0; j < 10; j++) { s[j] = __expf(s[j] - mx); sum += s[j]; }
    float inv = 1.f / sum;
    for (int j = 0; j < 10; j++) SC[r * 20 + h * 10 + j] = s[j] * inv;
  }
  __syncthreads();
  for (int idx = tid; idx < 64 * 256; idx += 256) QF[idx] = 0.f;
  __syncthreads();
  // pass 2: ctx += attn * v_j
  for (int j = 0; j < 10; j++) {
    for (int idx = tid; idx < 64 * 512; idx += 256) {
      int r = idx >> 9, c = idx & 511;
      KA[idx] = (c < 256) ? (bf16_t)upd[(size_t)NB[r * 10 + j] * 256 + c]
                          : (bf16_t)__cosf(DTT[r * 10 + j] * twf(c - 256));
    }
    __syncthreads();
    for (int nc = 0; nc < 256; nc += 32) {
      stage_wt(Wv, 256, nc, 512, WT, tid); __syncthreads();
      v8f acc = {};
      for (int k = 0; k < 512; k += 32)
        acc = wmma_bf16(frag_a(KA, 512, rt, k, lane), frag_b(WT, 512, ctl, k, lane), acc);
      int col = nc + ctl + (lane & 15); int mb = rt + 8 * (lane >> 4);
      int head = col >> 7;
#pragma unroll
      for (int i = 0; i < 8; i++) {
        float a = SC[(mb + i) * 20 + head * 10 + j];
        QF[(mb + i) * 256 + col] += a * acc[i];
      }
      __syncthreads();
    }
  }
  // merge: A = [ctx | node_feat]
  for (int idx = tid; idx < 64 * 512; idx += 256) {
    int r = idx >> 9, c = idx & 511;
    KA[idx] = (c < 256) ? (bf16_t)QF[r * 256 + c] : (bf16_t)NF[r * 256 + (c - 256)];
  }
  __syncthreads();
  for (int nc = 0; nc < 256; nc += 32) {
    stage_wt(mW1, 256, nc, 512, WT, tid); __syncthreads();
    v8f acc = {};
    for (int k = 0; k < 512; k += 32)
      acc = wmma_bf16(frag_a(KA, 512, rt, k, lane), frag_b(WT, 512, ctl, k, lane), acc);
    int col = nc + ctl + (lane & 15); float bb = mb1[col]; int mb = rt + 8 * (lane >> 4);
#pragma unroll
    for (int i = 0; i < 8; i++) {
      float v = acc[i] + bb;
      HM[(mb + i) * 256 + col] = (bf16_t)(v > 0.f ? v : 0.f);
    }
    __syncthreads();
  }
  for (int nc = 0; nc < 256; nc += 32) {
    stage_wt(mW2, 256, nc, 256, WT, tid); __syncthreads();
    v8f acc = {};
    for (int k = 0; k < 256; k += 32)
      acc = wmma_bf16(frag_a(HM, 256, rt, k, lane), frag_b(WT, 256, ctl, k, lane), acc);
    int col = nc + ctl + (lane & 15); float bb = mb2[col]; int mb = rt + 8 * (lane >> 4);
#pragma unroll
    for (int i = 0; i < 8; i++) {
      int g = g0 + mb + i;
      if (g < twoB) emb[(size_t)g * 256 + col] = acc[i] + bb;
    }
    __syncthreads();
  }
}

// =====================================================================
// Kernel 4: per-row normalize + row_cos outputs + normalized bf16 operands
// =====================================================================
__launch_bounds__(256)
__global__ void k_rowstats(const int* __restrict__ srcn, const int* __restrict__ dstn,
                           const float* __restrict__ emb, const float* __restrict__ upd,
                           const float* __restrict__ mem, bf16_t* __restrict__ an,
                           float* __restrict__ out, int B) {
  int wid = (blockIdx.x * blockDim.x + threadIdx.x) >> 5;
  int lane = threadIdx.x & 31;
  if (wid >= B) return;
  int s = srcn[wid], d = dstn[wid];
  const float* pe = emb + (size_t)wid * 256;
  const float* de = emb + (size_t)(B + wid) * 256;
  const float* nm = upd + (size_t)s * 256;
  const float* pm = mem + (size_t)s * 256;
  const float* dn = upd + (size_t)d * 256;
  const float* dm = mem + (size_t)d * 256;
  float v_pe[8], v_de[8], v_nm[8], v_pm[8], v_dn[8], v_dm[8];
  float s_pe = 0, s_de = 0, s_nm = 0, s_pm = 0, s_dn = 0, s_dm = 0;
  float d0 = 0, d1 = 0, d2 = 0, d3 = 0;
#pragma unroll
  for (int i = 0; i < 8; i++) {
    int c = lane + 32 * i;
    v_pe[i] = pe[c]; v_de[i] = de[c]; v_nm[i] = nm[c];
    v_pm[i] = pm[c]; v_dn[i] = dn[c]; v_dm[i] = dm[c];
    s_pe += v_pe[i] * v_pe[i]; s_de += v_de[i] * v_de[i]; s_nm += v_nm[i] * v_nm[i];
    s_pm += v_pm[i] * v_pm[i]; s_dn += v_dn[i] * v_dn[i]; s_dm += v_dm[i] * v_dm[i];
    d0 += v_pe[i] * v_nm[i]; d1 += v_nm[i] * v_pm[i];
    d2 += v_de[i] * v_dn[i]; d3 += v_dn[i] * v_dm[i];
  }
#pragma unroll
  for (int m = 1; m < 32; m <<= 1) {
    s_pe += __shfl_xor(s_pe, m, 32); s_de += __shfl_xor(s_de, m, 32);
    s_nm += __shfl_xor(s_nm, m, 32); s_pm += __shfl_xor(s_pm, m, 32);
    s_dn += __shfl_xor(s_dn, m, 32); s_dm += __shfl_xor(s_dm, m, 32);
    d0 += __shfl_xor(d0, m, 32); d1 += __shfl_xor(d1, m, 32);
    d2 += __shfl_xor(d2, m, 32); d3 += __shfl_xor(d3, m, 32);
  }
  float n_pe = sqrtf(s_pe) + 1e-8f, n_de = sqrtf(s_de) + 1e-8f;
  float n_nm = sqrtf(s_nm) + 1e-8f, n_pm = sqrtf(s_pm) + 1e-8f;
  float n_dn = sqrtf(s_dn) + 1e-8f, n_dm = sqrtf(s_dm) + 1e-8f;
  if (lane == 0) {
    out[wid]         = __expf(d0 / (n_pe * n_nm));  // pos_rec
    out[B + wid]     = __expf(d1 / (n_nm * n_pm));  // pos_drift
    out[2 * B + wid] = __expf(d2 / (n_de * n_dn));  // dst_pos_rec
    out[3 * B + wid] = __expf(d3 / (n_dn * n_dm));  // dst_pos_drift
  }
  bf16_t* a0 = an;
  bf16_t* a1 = an + (size_t)B * 256;
  bf16_t* a2 = an + (size_t)2 * B * 256;
  bf16_t* a3 = an + (size_t)3 * B * 256;
#pragma unroll
  for (int i = 0; i < 8; i++) {
    int c = lane + 32 * i;
    a0[(size_t)wid * 256 + c] = (bf16_t)(v_pe[i] / n_pe);
    a1[(size_t)wid * 256 + c] = (bf16_t)(v_nm[i] / n_nm);
    a2[(size_t)wid * 256 + c] = (bf16_t)(v_de[i] / n_de);
    a3[(size_t)wid * 256 + c] = (bf16_t)(v_dn[i] / n_dn);
  }
}

// =====================================================================
// Kernel 5: negatives = mem_after[seen] (mark ? upd : mem), normalized bf16
// =====================================================================
__launch_bounds__(256)
__global__ void k_negN(const int* __restrict__ seen, const int* __restrict__ mark,
                       const float* __restrict__ upd, const float* __restrict__ mem,
                       bf16_t* __restrict__ negN, int B) {
  int wid = (blockIdx.x * blockDim.x + threadIdx.x) >> 5;
  int lane = threadIdx.x & 31;
  if (wid >= B) return;
  int n = seen[wid];
  const float* srcp = mark[n] ? (upd + (size_t)n * 256) : (mem + (size_t)n * 256);
  float v[8]; float ss = 0.f;
#pragma unroll
  for (int i = 0; i < 8; i++) { int c = lane + 32 * i; v[i] = srcp[c]; ss += v[i] * v[i]; }
  for (int m = 1; m < 32; m <<= 1) ss += __shfl_xor(ss, m, 32);
  float nn = sqrtf(ss) + 1e-8f;
#pragma unroll
  for (int i = 0; i < 8; i++) negN[(size_t)wid * 256 + lane + 32 * i] = (bf16_t)(v[i] / nn);
}

// =====================================================================
// Kernel 6: neg sums: negsum[a][row] = sum_j exp(an_a[row] . negN[j])
// A tile is a contiguous 32KB bf16 region -> CDNA5 async global->LDS copy.
// =====================================================================
__launch_bounds__(256)
__global__ void k_cross(const bf16_t* __restrict__ an, const bf16_t* __restrict__ negN,
                        float* __restrict__ negsum, int B) {
  extern __shared__ char smem[];
  bf16_t* A = (bf16_t*)smem;           // 64x256
  float* rs = (float*)(A + 64 * 256);  // 64 row partials
  const int tid = threadIdx.x, lane = tid & 31, wv = tid >> 5;
  const int rb = blockIdx.x * 64;
  const int ai = blockIdx.y;
  const int cb = blockIdx.z * 1024;
  const bf16_t* Aa = an + (size_t)ai * B * 256 + (size_t)rb * 256;
  // async stage: 2048 x 16B ops, 8 per thread
  for (int idx = tid; idx < (64 * 256) / 8; idx += 256)
    async_copy_b128((void*)(A + idx * 8), (const void*)(Aa + (size_t)idx * 8));
  if (tid < 64) rs[tid] = 0.f;
  async_wait0();
  __syncthreads();
  for (int t = wv; t < 256; t += 8) {
    int rt = (t >> 6) * 16;
    int n0 = cb + (t & 63) * 16;   // negN row-major IS the transposed-B layout
    // prefetch next panel of negN (lowers to global_prefetch_b8)
    if (t + 8 < 256) {
      int n1 = cb + ((t + 8) & 63) * 16;
      __builtin_prefetch(negN + (size_t)(n1 + (lane & 15)) * 256, 0, 1);
    }
    v8f acc = {};
    for (int k = 0; k < 256; k += 32)
      acc = wmma_bf16(frag_a(A, 256, rt, k, lane), frag_b(negN, 256, n0, k, lane), acc);
    int mb = rt + 8 * (lane >> 4);
#pragma unroll
    for (int i = 0; i < 8; i++) {
      float e = __expf(acc[i]);
      e += __shfl_xor(e, 1, 16); e += __shfl_xor(e, 2, 16);
      e += __shfl_xor(e, 4, 16); e += __shfl_xor(e, 8, 16);
      if ((lane & 15) == 0) atomicAdd(&rs[mb + i], e);
    }
  }
  __syncthreads();
  if (tid < 64) atomicAdd(&negsum[(size_t)ai * B + rb + tid], rs[tid]);
}

// =====================================================================
// Kernels 7-10: segment counts/max + checksum + loss
// =====================================================================
__global__ void k_cnt(const int* __restrict__ s, const int* __restrict__ d,
                      const float* __restrict__ et, float* cnt_s, float* cnt_d,
                      float* ts_s, float* ts_d, int B) {
  int i = blockIdx.x * blockDim.x + threadIdx.x;
  if (i >= B) return;
  float t = et[i];
  atomicAdd(&cnt_s[s[i]], 1.0f);
  atomicAdd(&cnt_d[d[i]], 1.0f);
  atomicMax((unsigned int*)&ts_s[s[i]], __float_as_uint(t)); // t >= 0
  atomicMax((unsigned int*)&ts_d[d[i]], __float_as_uint(t));
}

// sum_n sum_dims mean-message == sum_edges rowsum(m_e)/cnt[src_e]
__launch_bounds__(256)
__global__ void k_msgrow(const int* __restrict__ s, const int* __restrict__ d,
                         const float* __restrict__ et, const float* __restrict__ lastup,
                         const float* __restrict__ upd, const float* __restrict__ cnt_s,
                         const float* __restrict__ cnt_d, float* cks, int B) {
  int wid = (blockIdx.x * blockDim.x + threadIdx.x) >> 5;
  int lane = threadIdx.x & 31;
  if (wid >= 2 * B) return;
  int e = wid >> 1, side = wid & 1;
  int sn = side ? d[e] : s[e];
  int on = side ? s[e] : d[e];
  float dte = et[e] - lastup[sn];
  float sum = 0.f;
#pragma unroll
  for (int i = 0; i < 16; i++) {
    int c = lane + 32 * i;
    sum += (c < 256) ? upd[(size_t)on * 256 + c] : __cosf(dte * twf(c - 256));
  }
  for (int m = 1; m < 32; m <<= 1) sum += __shfl_xor(sum, m, 32);
  if (lane == 0) {
    float cnt = side ? cnt_d[sn] : cnt_s[sn];
    atomicAdd(cks, sum / fmaxf(cnt, 1.0f));
  }
}

__global__ void k_tssum(const float* cnt_s, const float* cnt_d,
                        const float* ts_s, const float* ts_d, float* cks, int Nn) {
  __shared__ float red[256];
  int i = blockIdx.x * blockDim.x + threadIdx.x;
  float v = 0.f;
  if (i < Nn) {
    if (cnt_s[i] > 0.f) v += ts_s[i];
    if (cnt_d[i] > 0.f) v += ts_d[i];
  }
  red[threadIdx.x] = v; __syncthreads();
  for (int st = 128; st > 0; st >>= 1) {
    if (threadIdx.x < st) red[threadIdx.x] += red[threadIdx.x + st];
    __syncthreads();
  }
  if (threadIdx.x == 0) atomicAdd(cks, red[0]);
}

__global__ void k_loss(const float* __restrict__ out, const float* __restrict__ negsum,
                       float* lacc, int B) {
  __shared__ float red[256];
  int i = blockIdx.x * blockDim.x + threadIdx.x;
  float v = 0.f;
  if (i < B) {
    float pr = out[i], pd = out[B + i], dpr = out[2 * B + i], dpd = out[3 * B + i];
    float nr = negsum[i], nd = negsum[B + i], dnr = negsum[2 * B + i], dnd = negsum[3 * B + i];
    v = -__logf(pr / nr) - __logf(dpr / dnr) - __logf(pd / nd) - __logf(dpd / dnd);
  }
  red[threadIdx.x] = v; __syncthreads();
  for (int st = 128; st > 0; st >>= 1) {
    if (threadIdx.x < st) red[threadIdx.x] += red[threadIdx.x + st];
    __syncthreads();
  }
  if (threadIdx.x == 0) atomicAdd(lacc, red[0]);
}

__global__ void k_scalars(const float* lacc, const float* cks, float* out, int B) {
  if (threadIdx.x == 0) {
    out[4 * B] = lacc[0] / (float)B;
    out[4 * B + 1] = cks[0];
  }
}

__global__ void k_zero(int* p, long n) {
  long i = blockIdx.x * (long)blockDim.x + threadIdx.x;
  if (i < n) p[i] = 0;
}

// =====================================================================
extern "C" void kernel_launch(void* const* d_in, const int* in_sizes, int n_in,
                              void* d_out, int out_size, void* d_ws, size_t ws_size,
                              hipStream_t stream) {
  const int*   srcn  = (const int*)d_in[0];
  const int*   dstn  = (const int*)d_in[1];
  const int*   seen  = (const int*)d_in[2];
  const int*   snb   = (const int*)d_in[3];
  const int*   dnb   = (const int*)d_in[4];
  const float* et    = (const float*)d_in[5];
  const float* snbt  = (const float*)d_in[6];
  const float* dnbt  = (const float*)d_in[7];
  const float* mem   = (const float*)d_in[8];
  const float* lastu = (const float*)d_in[9];
  const float* msgs  = (const float*)d_in[10];
  const float* W1    = (const float*)d_in[11];
  const float* b1    = (const float*)d_in[12];
  const float* W2    = (const float*)d_in[13];
  const float* b2    = (const float*)d_in[14];
  const float* Wih   = (const float*)d_in[15];
  const float* Whh   = (const float*)d_in[16];
  const float* bih   = (const float*)d_in[17];
  const float* bhh   = (const float*)d_in[18];
  const float* Wq    = (const float*)d_in[19];
  const float* Wk    = (const float*)d_in[20];
  const float* Wv    = (const float*)d_in[21];
  const float* mW1   = (const float*)d_in[22];
  const float* mb1   = (const float*)d_in[23];
  const float* mW2   = (const float*)d_in[24];
  const float* mb2   = (const float*)d_in[25];
  float* outF = (float*)d_out;

  const int B  = in_sizes[0];
  const int Nn = in_sizes[9];

  // ---- workspace carve-up ----
  size_t off = 0;
  auto alloc = [&](size_t bytes) { size_t o = off; off = (off + bytes + 255) & ~(size_t)255; return o; };
  float*  upd  = (float*)((char*)d_ws + alloc((size_t)Nn * 256 * 4));
  float*  emb  = (float*)((char*)d_ws + alloc((size_t)2 * B * 256 * 4));
  bf16_t* an   = (bf16_t*)((char*)d_ws + alloc((size_t)4 * B * 256 * 2));
  bf16_t* negN = (bf16_t*)((char*)d_ws + alloc((size_t)B * 256 * 2));
  size_t zwords = 5 * (size_t)Nn + 4 * (size_t)B + 2;
  int*   zbase  = (int*)((char*)d_ws + alloc(zwords * 4));
  int*   mark   = zbase;
  float* cnt_s  = (float*)(zbase + Nn);
  float* cnt_d  = cnt_s + Nn;
  float* ts_s   = cnt_d + Nn;
  float* ts_d   = ts_s + Nn;
  float* negsum = ts_d + Nn;
  float* lacc   = negsum + 4 * (size_t)B;
  float* cks    = lacc + 1;

  const size_t LDS1 = (size_t)(64 * 512 + 64 * 256 + 64 * 128 + 64 * 256 + 32 * 512) * 2;
  const size_t LDS2 = (size_t)(64 * 512 + 64 * 256 + 32 * 512) * 2 +
                      (size_t)(64 * 256 + 64 * 256 + 64 * 20 + 64 * 10 + 64) * 4 +
                      (size_t)(64 * 10 + 64) * 4;
  const size_t LDS3 = (size_t)(64 * 256) * 2 + 64 * 4;

  k_zero<<<(int)((zwords + 255) / 256), 256, 0, stream>>>(zbase, (long)zwords);

  k_node_update<<<(Nn + 63) / 64, 256, LDS1, stream>>>(
      msgs, mem, W1, b1, W2, b2, Wih, Whh, bih, bhh, upd, Nn);

  k_mark<<<(2 * B + 255) / 256, 256, 0, stream>>>(srcn, dstn, mark, B);

  k_attention<<<(2 * B + 63) / 64, 256, LDS2, stream>>>(
      srcn, dstn, snb, dnb, et, snbt, dnbt, upd, Wq, Wk, Wv, mW1, mb1, mW2, mb2, emb, B);

  k_rowstats<<<(B * 32 + 255) / 256, 256, 0, stream>>>(srcn, dstn, emb, upd, mem, an, outF, B);
  k_negN<<<(B * 32 + 255) / 256, 256, 0, stream>>>(seen, mark, upd, mem, negN, B);

  dim3 gc((B + 63) / 64, 4, (B + 1023) / 1024);
  k_cross<<<gc, 256, LDS3, stream>>>(an, negN, negsum, B);

  k_cnt<<<(B + 255) / 256, 256, 0, stream>>>(srcn, dstn, et, cnt_s, cnt_d, ts_s, ts_d, B);
  k_msgrow<<<(2 * B * 32 + 255) / 256, 256, 0, stream>>>(
      srcn, dstn, et, lastu, upd, cnt_s, cnt_d, cks, B);
  k_tssum<<<(Nn + 255) / 256, 256, 0, stream>>>(cnt_s, cnt_d, ts_s, ts_d, cks, Nn);

  k_loss<<<(B + 255) / 256, 256, 0, stream>>>(outF, negsum, lacc, B);
  k_scalars<<<1, 32, 0, stream>>>(lacc, cks, outF, B);

  (void)n_in; (void)out_size; (void)ws_size;
}
